// TrueGroupedQueryAttention_84928683311124
// MI455X (gfx1250) — matrix operand order repaired and checked by reading, not compiled, not run on hardware
//
#include <hip/hip_runtime.h>
#include <stdint.h>
#include <stddef.h>

// Problem constants (match reference)
#define BB   2
#define SS   2048
#define DD   1024
#define HH   16
#define GG   4
#define HD   64
#define MM   (BB * SS)      // 4096 token rows
#define DQ   (GG * HD)      // 256 q-projection width

// LDS row strides (elements). 16B-aligned and bank-conflict-free for
// "16 rows at same column" fragment reads: 40*2B=80B=20 dwords, 72*2B=144B=36
// dwords; both give 16 distinct banks over 16 consecutive rows.
#define GST  40
#define AST  72

typedef __bf16 bf16;
typedef __attribute__((ext_vector_type(16))) __bf16 v16bf;
typedef __attribute__((ext_vector_type(8)))  float  v8f;

// 32-byte fragment container: two 16-byte loads -> one v16bf WMMA operand.
union V16 { v16bf v; uint4 q[2]; };
union V8U { uint4 u; bf16 e[8]; };

__device__ __forceinline__ bf16 f2bf(float f) {
  unsigned u = __float_as_uint(f);
  unsigned r = (u + 0x7fffu + ((u >> 16) & 1u)) >> 16;  // RNE
  return __builtin_bit_cast(bf16, (unsigned short)r);
}
__device__ __forceinline__ float bf2f(bf16 b) {
  unsigned short s = __builtin_bit_cast(unsigned short, b);
  return __uint_as_float(((unsigned)s) << 16);
}

// CDNA5 async global->LDS copy (ASYNCcnt-tracked, bypasses VGPRs).
// lds_off: byte offset inside the workgroup's LDS allocation (low 32 bits of a
// generic pointer to a __shared__ object are exactly this offset on gfx1250).
__device__ __forceinline__ void async_load_b128(unsigned lds_off, const void* gptr) {
  asm volatile("global_load_async_to_lds_b128 %0, %1, off"
               :: "v"(lds_off), "v"((unsigned long long)gptr)
               : "memory");
}
__device__ __forceinline__ void wait_async0() {
  asm volatile("s_wait_asynccnt 0" ::: "memory");
}
__device__ __forceinline__ unsigned lds_off(const void* p) {
  return (unsigned)(unsigned long long)p;
}

// ---------------------------------------------------------------- converts
__global__ void cvt_f32_bf16(const float* __restrict__ in, bf16* __restrict__ out, int n) {
  int idx = blockIdx.x * blockDim.x + threadIdx.x;
  if (idx < n) out[idx] = f2bf(in[idx]);
}

// out[n*K + k] = in[k*N + n]  (weights stored (N,K) so B-fragments read contiguous K)
__global__ void transpose_cvt(const float* __restrict__ in, bf16* __restrict__ out, int K, int N) {
  int idx = blockIdx.x * blockDim.x + threadIdx.x;
  if (idx >= K * N) return;
  int kk = idx / N, nn = idx % N;
  out[(size_t)nn * K + kk] = f2bf(in[idx]);
}

// ---------------------------------------------------------------- GEMM
// One K-step of the wave's 64x32 output slab: 4 A-frags, 2 B-frags, 8 WMMAs.
__device__ __forceinline__ void gemm_compute_tile(
    const bf16 (*la)[GST], const bf16 (*lb)[GST],
    int wm, int wn, int lmod, int lhalf, v8f acc[4][2])
{
  V16 af[4], bfg[2];
#pragma unroll
  for (int i = 0; i < 4; ++i) {
    const bf16* p = &la[wm * 64 + i * 16 + lmod][0];
    af[i].q[0] = *(const uint4*)(p + 16 * lhalf);       // K = 16*lhalf .. +7
    af[i].q[1] = *(const uint4*)(p + 16 * lhalf + 8);   // K = 16*lhalf+8 .. +15
  }
#pragma unroll
  for (int j = 0; j < 2; ++j) {
    const bf16* p = &lb[wn * 32 + j * 16 + lmod][0];
    bfg[j].q[0] = *(const uint4*)(p + 16 * lhalf);      // K = 16*lhalf .. +7
    bfg[j].q[1] = *(const uint4*)(p + 16 * lhalf + 8);  // K = 16*lhalf+8 .. +15
  }
#pragma unroll
  for (int i = 0; i < 4; ++i)
#pragma unroll
    for (int j = 0; j < 2; ++j)
      acc[i][j] = __builtin_amdgcn_wmma_f32_16x16x32_bf16(
          false, af[i].v, false, bfg[j].v, (short)0, acc[i][j], false, false);
}

// C(M,N) = A(M,K) @ B(K,N) + bias ;  A bf16 row-major, B given transposed (N,K).
// Block tile 128x128, K-step 32, 8 waves (2x4), each wave 64x32 (4x2 WMMA tiles).
// Double-buffered LDS staged with async global->LDS loads; pipeline unrolled by
// 2 so buffer indices are compile-time constants (no cndmask/branches in loop).
// Requires Kdim to be a multiple of 64 (true for all call sites: K = 1024).
__global__ __launch_bounds__(256) void gemm_bf16(
    const bf16* __restrict__ A, const bf16* __restrict__ Bt,
    const float* __restrict__ bias,
    bf16* __restrict__ out_bf, float* __restrict__ out_f32,
    int Mdim, int Ndim, int Kdim)
{
  __shared__ __align__(16) bf16 lA[2][128][GST];
  __shared__ __align__(16) bf16 lB[2][128][GST];

  const int tid   = threadIdx.x;
  const int lane  = tid & 31;
  const int wave  = tid >> 5;
  const int lmod  = lane & 15;
  const int lhalf = lane >> 4;
  const int wm = wave & 1;          // 0..1 -> 64-row slab
  const int wn = wave >> 1;         // 0..3 -> 32-col slab
  const int m0 = blockIdx.x * 128;
  const int n0 = blockIdx.y * 128;

  v8f acc[4][2];
  for (int i = 0; i < 4; ++i)
    for (int j = 0; j < 2; ++j)
      for (int r = 0; r < 8; ++r) acc[i][j][r] = 0.0f;

  // Each thread owns one 16B segment of the A tile and one of the B tile.
  const int ldrow  = tid >> 1;
  const int ldhalf = (tid & 1) * 16;
  const bf16* aptr = A  + (size_t)(m0 + ldrow) * Kdim + ldhalf;
  const bf16* bptr = Bt + (size_t)(n0 + ldrow) * Kdim + ldhalf;
  unsigned aoff[2], boff[2];
  aoff[0] = lds_off(&lA[0][ldrow][ldhalf]);
  aoff[1] = lds_off(&lA[1][ldrow][ldhalf]);
  boff[0] = lds_off(&lB[0][ldrow][ldhalf]);
  boff[1] = lds_off(&lB[1][ldrow][ldhalf]);

  const int T = Kdim >> 5;          // #K-tiles (even)

  // Prologue: stage tile 0 into buf 0.
  async_load_b128(aoff[0], aptr);
  async_load_b128(boff[0], bptr);

  int kt = 0;
  for (int t = 0; t + 2 < T; t += 2) {
    wait_async0();                       // my pieces of tile(t) landed
    __syncthreads();                     // everyone's landed; buf1 writable
    async_load_b128(aoff[1], aptr + kt + 32);
    async_load_b128(boff[1], bptr + kt + 32);
    gemm_compute_tile(lA[0], lB[0], wm, wn, lmod, lhalf, acc);

    wait_async0();
    __syncthreads();
    async_load_b128(aoff[0], aptr + kt + 64);
    async_load_b128(boff[0], bptr + kt + 64);
    gemm_compute_tile(lA[1], lB[1], wm, wn, lmod, lhalf, acc);
    kt += 64;
  }
  // Epilogue pair: tiles T-2 (buf0) and T-1 (buf1).
  wait_async0();
  __syncthreads();
  async_load_b128(aoff[1], aptr + kt + 32);
  async_load_b128(boff[1], bptr + kt + 32);
  gemm_compute_tile(lA[0], lB[0], wm, wn, lmod, lhalf, acc);
  wait_async0();
  __syncthreads();
  gemm_compute_tile(lA[1], lB[1], wm, wn, lmod, lhalf, acc);

#pragma unroll
  for (int i = 0; i < 4; ++i)
#pragma unroll
    for (int j = 0; j < 2; ++j) {
      const int col = n0 + wn * 32 + j * 16 + lmod;
      const float bv = bias[col];
#pragma unroll
      for (int r = 0; r < 8; ++r) {
        const int row = m0 + wm * 64 + i * 16 + r + 8 * lhalf;
        const float v = acc[i][j][r] + bv;
        if (out_bf) out_bf[(size_t)row * Ndim + col] = f2bf(v);
        else        out_f32[(size_t)row * Ndim + col] = v;
      }
    }
}

// ---------------------------------------------------------------- RoPE (in place, bf16)
__global__ void rope_kernel(bf16* __restrict__ data, int nheads, int total) {
  int idx = blockIdx.x * blockDim.x + threadIdx.x;
  if (idx >= total) return;
  const int i   = idx & 31;                 // rotation pair index 0..31
  const int h   = (idx >> 5) % nheads;
  const int row = idx / (32 * nheads);      // b*S + s
  const int pos = row & (SS - 1);
  const float inv = powf(10000.0f, -(float)(2 * i) / 64.0f);
  const float f = (float)pos * inv;
  float s, c;
  sincosf(f, &s, &c);
  bf16* p = data + ((size_t)row * nheads + h) * HD;
  const float x1 = bf2f(p[i]);
  const float x2 = bf2f(p[i + 32]);
  p[i]      = f2bf(x1 * c - x2 * s);
  p[i + 32] = f2bf(x2 * c + x1 * s);
}

// ---------------------------------------------------------------- attention
// grid (S/64, B*H); 128 threads = 4 waves; wave owns 16 q rows, block owns 64.
// q head g = h>>2 is shared by kv head h (reference's (g,j) -> h = g*4+j).
__global__ __launch_bounds__(128) void attn_kernel(
    const bf16* __restrict__ q,   // (B*S, 4, 64)
    const bf16* __restrict__ k,   // (B*S, 16, 64)
    const bf16* __restrict__ v,   // (B*S, 16, 64)
    bf16* __restrict__ o)         // (B*S, 16, 64)
{
  __shared__ __align__(16) bf16 kt[64][AST];       // (key, hd)  -> score B frags
  __shared__ __align__(16) bf16 vt[64][AST];       // (hd, key)  -> PV B frags
  __shared__ __align__(16) bf16 pt[4][16][AST];    // per-wave P (qrow, key)

  const int qblk = blockIdx.x;
  const int bh   = blockIdx.y;
  const int b = bh >> 4, h = bh & 15, g = h >> 2;
  const int tid = threadIdx.x, lane = tid & 31, wave = tid >> 5;
  const int lmod = lane & 15, lhalf = lane >> 4;
  const int qrow0 = qblk * 64 + wave * 16;

  // Preload this wave's Q tile (16 x 64) as two K=32 A-fragments, from global.
  V16 qf[2];
  {
    const bf16* qrow = q + ((size_t)(b * SS) + qrow0 + lmod) * DQ + g * HD;
#pragma unroll
    for (int c = 0; c < 2; ++c) {
      qf[c].q[0] = *(const uint4*)(qrow + 32 * c + 16 * lhalf);
      qf[c].q[1] = *(const uint4*)(qrow + 32 * c + 16 * lhalf + 8);
    }
  }

  v8f oacc[4];
  for (int jn = 0; jn < 4; ++jn)
    for (int r = 0; r < 8; ++r) oacc[jn][r] = 0.0f;
  float m_i[8], l_i[8];
  for (int r = 0; r < 8; ++r) { m_i[r] = -3.0e38f; l_i[r] = 0.0f; }

  const size_t kvbase = (size_t)(b * SS) * (HH * HD) + (size_t)h * HD;

  for (int kb = 0; kb <= qblk; ++kb) {
    // K tile: async DMA straight into LDS. V tile: load + transpose-scatter.
#pragma unroll
    for (int it = 0; it < 4; ++it) {
      const int e   = tid + it * 128;       // 0..511
      const int key = e >> 3;
      const int seg = (e & 7) * 8;
      const size_t rowoff = kvbase + (size_t)(kb * 64 + key) * (HH * HD) + seg;
      async_load_b128(lds_off(&kt[key][seg]), k + rowoff);
      V8U vv; vv.u = *(const uint4*)(v + rowoff);
#pragma unroll
      for (int x = 0; x < 8; ++x) vt[seg + x][key] = vv.e[x];
    }
    wait_async0();
    __syncthreads();

    // S = Q @ K^T  (16 q-rows x 64 keys = 4 WMMA tiles, K-dim = 64 = 2 chunks)
    v8f sc[4];
    for (int j = 0; j < 4; ++j)
      for (int r = 0; r < 8; ++r) sc[j][r] = 0.0f;
#pragma unroll
    for (int j = 0; j < 4; ++j) {
      const bf16* kp = &kt[j * 16 + lmod][0];
#pragma unroll
      for (int c = 0; c < 2; ++c) {
        V16 kf;
        kf.q[0] = *(const uint4*)(kp + 32 * c + 16 * lhalf);
        kf.q[1] = *(const uint4*)(kp + 32 * c + 16 * lhalf + 8);
        sc[j] = __builtin_amdgcn_wmma_f32_16x16x32_bf16(
            false, qf[c].v, false, kf.v, (short)0, sc[j], false, false);
      }
    }

    // scale + causal mask + per-row max (row = r + 8*lhalf, keys across 16 lanes)
    float rowmax[8];
    for (int r = 0; r < 8; ++r) rowmax[r] = -3.0e38f;
#pragma unroll
    for (int j = 0; j < 4; ++j)
#pragma unroll
      for (int r = 0; r < 8; ++r) {
        const int kk = kb * 64 + j * 16 + lmod;
        const int qr = qrow0 + r + 8 * lhalf;
        float sval = (kk <= qr) ? sc[j][r] * 0.125f : -3.0e38f;
        sc[j][r] = sval;
        rowmax[r] = fmaxf(rowmax[r], sval);
      }
#pragma unroll
    for (int r = 0; r < 8; ++r)
      for (int d = 1; d < 16; d <<= 1)
        rowmax[r] = fmaxf(rowmax[r], __shfl_xor(rowmax[r], d, 32));

    // online softmax rescale
#pragma unroll
    for (int r = 0; r < 8; ++r) {
      const float mnew = fmaxf(m_i[r], rowmax[r]);
      const float corr = __expf(m_i[r] - mnew);
      m_i[r] = mnew;
      l_i[r] *= corr;
      for (int jn = 0; jn < 4; ++jn) oacc[jn][r] *= corr;
    }
    float rowsum[8];
    for (int r = 0; r < 8; ++r) rowsum[r] = 0.0f;
#pragma unroll
    for (int j = 0; j < 4; ++j)
#pragma unroll
      for (int r = 0; r < 8; ++r) {
        const float p = __expf(sc[j][r] - m_i[r]);
        sc[j][r] = p;
        rowsum[r] += p;
      }
#pragma unroll
    for (int r = 0; r < 8; ++r) {
      for (int d = 1; d < 16; d <<= 1) rowsum[r] += __shfl_xor(rowsum[r], d, 32);
      l_i[r] += rowsum[r];
    }

    // Stage P (C-layout) -> per-wave LDS -> A-layout fragments.
#pragma unroll
    for (int j = 0; j < 4; ++j)
#pragma unroll
      for (int r = 0; r < 8; ++r)
        pt[wave][r + 8 * lhalf][j * 16 + lmod] = f2bf(sc[j][r]);
    asm volatile("s_wait_dscnt 0" ::: "memory");   // same-wave LDS RAW fence

    // O += P @ V
#pragma unroll
    for (int c = 0; c < 2; ++c) {
      V16 pf;
      const bf16* pp = &pt[wave][lmod][0];
      pf.q[0] = *(const uint4*)(pp + 32 * c + 16 * lhalf);
      pf.q[1] = *(const uint4*)(pp + 32 * c + 16 * lhalf + 8);
#pragma unroll
      for (int jn = 0; jn < 4; ++jn) {
        V16 vf;
        const bf16* vp = &vt[jn * 16 + lmod][0];
        vf.q[0] = *(const uint4*)(vp + 32 * c + 16 * lhalf);
        vf.q[1] = *(const uint4*)(vp + 32 * c + 16 * lhalf + 8);
        oacc[jn] = __builtin_amdgcn_wmma_f32_16x16x32_bf16(
            false, pf.v, false, vf.v, (short)0, oacc[jn], false, false);
      }
    }
    __syncthreads();   // before kt/vt are overwritten
  }

  // normalize + store (b, s, h, hd) = (b, s, D) head-concat order
#pragma unroll
  for (int jn = 0; jn < 4; ++jn)
#pragma unroll
    for (int r = 0; r < 8; ++r) {
      const int qr = qrow0 + r + 8 * lhalf;
      const int hd = jn * 16 + lmod;
      const float val = oacc[jn][r] / l_i[r];
      o[((size_t)(b * SS) + qr) * (HH * HD) + (size_t)h * HD + hd] = f2bf(val);
    }
}

// ---------------------------------------------------------------- launch
extern "C" void kernel_launch(void* const* d_in, const int* in_sizes, int n_in,
                              void* d_out, int out_size, void* d_ws, size_t ws_size,
                              hipStream_t stream) {
  (void)in_sizes; (void)n_in; (void)out_size; (void)ws_size;
  const float* x  = (const float*)d_in[0];
  const float* Wq = (const float*)d_in[1];
  const float* bq = (const float*)d_in[2];
  const float* Wk = (const float*)d_in[3];
  const float* bk = (const float*)d_in[4];
  const float* Wv = (const float*)d_in[5];
  const float* bv = (const float*)d_in[6];
  const float* Wo = (const float*)d_in[7];
  const float* bo = (const float*)d_in[8];
  float* out = (float*)d_out;

  char* ws = (char*)d_ws;
  size_t off = 0;
  auto take = [&](size_t bytes) -> char* {
    char* p = ws + off;
    off += (bytes + 255) & ~(size_t)255;
    return p;
  };
  bf16* xb   = (bf16*)take((size_t)MM * DD * 2);   // x in bf16
  bf16* wqT  = (bf16*)take((size_t)DQ * DD * 2);   // Wq^T (N,K)
  bf16* wkT  = (bf16*)take((size_t)DD * DD * 2);
  bf16* wvT  = (bf16*)take((size_t)DD * DD * 2);
  bf16* woT  = (bf16*)take((size_t)DD * DD * 2);
  bf16* qb   = (bf16*)take((size_t)MM * DQ * 2);   // q (post-RoPE)
  bf16* kbuf = (bf16*)take((size_t)MM * DD * 2);   // k (post-RoPE)
  bf16* vbuf = (bf16*)take((size_t)MM * DD * 2);   // v
  bf16* ab   = (bf16*)take((size_t)MM * DD * 2);   // attention output

  // 1) precision conversion
  {
    const int n = MM * DD;
    cvt_f32_bf16<<<(n + 255) / 256, 256, 0, stream>>>(x, xb, n);
  }
  transpose_cvt<<<(DD * DQ + 255) / 256, 256, 0, stream>>>(Wq, wqT, DD, DQ);
  transpose_cvt<<<(DD * DD + 255) / 256, 256, 0, stream>>>(Wk, wkT, DD, DD);
  transpose_cvt<<<(DD * DD + 255) / 256, 256, 0, stream>>>(Wv, wvT, DD, DD);
  transpose_cvt<<<(DD * DD + 255) / 256, 256, 0, stream>>>(Wo, woT, DD, DD);

  // 2) Q/K/V projections (WMMA, async double-buffered)
  gemm_bf16<<<dim3(MM / 128, DQ / 128), 256, 0, stream>>>(xb, wqT, bq, qb,   nullptr, MM, DQ, DD);
  gemm_bf16<<<dim3(MM / 128, DD / 128), 256, 0, stream>>>(xb, wkT, bk, kbuf, nullptr, MM, DD, DD);
  gemm_bf16<<<dim3(MM / 128, DD / 128), 256, 0, stream>>>(xb, wvT, bv, vbuf, nullptr, MM, DD, DD);

  // 3) RoPE
  rope_kernel<<<(MM * GG * 32 + 255) / 256, 256, 0, stream>>>(qb,   GG, MM * GG * 32);
  rope_kernel<<<(MM * HH * 32 + 255) / 256, 256, 0, stream>>>(kbuf, HH, MM * HH * 32);

  // 4) flash attention (WMMA + async K staging)
  attn_kernel<<<dim3(SS / 64, BB * HH), 128, 0, stream>>>(qb, kbuf, vbuf, ab);

  // 5) output projection, fp32 result (WMMA)
  gemm_bf16<<<dim3(MM / 128, DD / 128), 256, 0, stream>>>(ab, woT, bo, nullptr, out, MM, DD, DD);
}
